// WEGAT_TOPK_Conv_17145509445847
// MI455X (gfx1250) — compile-verified, matrix operand order checked by reading, not run on hardware
//
#include <hip/hip_runtime.h>
#include <hip/hip_bf16.h>
#include <math.h>

#define N_NODES 50000
#define N_EDGES 800000
#define DIN     128
#define DOUT    64
#define EIN     16
#define EOUT    16
#define NHEAD   4
#define HC      (NHEAD * DOUT)   /* 256 */
#define NEG_SLOPE 0.2f

typedef __attribute__((ext_vector_type(16))) __bf16 v16bf;
typedef __attribute__((ext_vector_type(8)))  __bf16 v8bf;
typedef __attribute__((ext_vector_type(8)))  float  v8f;

// float -> bf16 round-to-nearest-even via bit manipulation (prep kernel only)
__device__ __forceinline__ __bf16 f2bf(float f) {
  union { float f; unsigned u; } v; v.f = f;
  unsigned r = (v.u + 0x7FFFu + ((v.u >> 16) & 1u)) >> 16;
  unsigned short hs = (unsigned short)r;
  union { unsigned short s; __bf16 b; } o; o.s = hs;
  return o.b;
}

__device__ __forceinline__ void atomicMaxF(float* addr, float val) {
  if (val >= 0.0f) atomicMax((int*)addr, __float_as_int(val));
  else             atomicMin((unsigned int*)addr, __float_as_uint(val));
}

// ---------------------------------------------------------------------------
// Kernel 0: re-initialize workspace state every call (graph-replay safe)
// ---------------------------------------------------------------------------
__global__ void k_init(float* __restrict__ m, float* __restrict__ s,
                       float* __restrict__ accum) {
  int i = blockIdx.x * blockDim.x + threadIdx.x;
  if (i < N_NODES * HC) accum[i] = 0.0f;
  if (i < N_NODES * NHEAD) { m[i] = -3.0e38f; s[i] = 0.0f; }
}

// ---------------------------------------------------------------------------
// Kernel 0b: one-shot fp32 -> bf16 of x and W_l (hoists cvt out of GEMM loop,
// halves GEMM read bandwidth)
// ---------------------------------------------------------------------------
#define XB_ELEMS (N_NODES * DIN)           /* 6,400,000 */
#define WB_ELEMS (HC * DIN)                /* 32,768   */
__global__ void k_prep_bf16(const float* __restrict__ x, const float* __restrict__ Wl,
                            __bf16* __restrict__ xb, __bf16* __restrict__ wb) {
  int i = blockIdx.x * blockDim.x + threadIdx.x;
  if (i < XB_ELEMS) xb[i] = f2bf(x[i]);
  else if (i < XB_ELEMS + WB_ELEMS) {
    int j = i - XB_ELEMS;
    wb[j] = f2bf(Wl[j]);
  }
}

// ---------------------------------------------------------------------------
// Kernel 1: xl = x @ W_l.T  via v_wmma_f32_16x16x32_bf16 on pre-converted bf16
// Block = 256 threads (8 waves) handles a 16-row strip; wave w does col-tiles
// {w, w+8}. A fragment reused across both WMMAs. Fragment halves are two
// contiguous 16-byte bf16 runs per the ISA 16-bit A/B VGPR layout.
// ---------------------------------------------------------------------------
__global__ __launch_bounds__(256)
void k_node_proj_wmma(const __bf16* __restrict__ xb, const __bf16* __restrict__ wb,
                      float* __restrict__ xl) {
  const int mb    = blockIdx.x;          // 16-row tile (N/16 == 3125 exact)
  const int wid   = threadIdx.x >> 5;    // wave 0..7
  const int lane  = threadIdx.x & 31;
  const int l16   = lane & 15;
  const int khalf = lane >> 4;           // K 0..7/16..23 vs 8..15/24..31

  const int row = mb * 16 + l16;
  const int nb0 = wid, nb1 = wid + 8;

  const __bf16* xr = xb + (size_t)row * DIN;
  const __bf16* w0 = wb + (size_t)(nb0 * 16 + l16) * DIN;  // B[k][j] = W_l[j][k]
  const __bf16* w1 = wb + (size_t)(nb1 * 16 + l16) * DIN;

  v8f acc0 = {}; v8f acc1 = {};
#pragma unroll
  for (int k0 = 0; k0 < DIN; k0 += 32) {
    const int base0 = k0 + khalf * 8;        // vector elems 0..7
    const int base1 = k0 + 16 + khalf * 8;   // vector elems 8..15
    v8bf alo = *(const v8bf*)(xr + base0);
    v8bf ahi = *(const v8bf*)(xr + base1);
    v8bf b0l = *(const v8bf*)(w0 + base0);
    v8bf b0h = *(const v8bf*)(w0 + base1);
    v8bf b1l = *(const v8bf*)(w1 + base0);
    v8bf b1h = *(const v8bf*)(w1 + base1);
    v16bf a  = __builtin_shufflevector(alo, ahi, 0,1,2,3,4,5,6,7,8,9,10,11,12,13,14,15);
    v16bf b0 = __builtin_shufflevector(b0l, b0h, 0,1,2,3,4,5,6,7,8,9,10,11,12,13,14,15);
    v16bf b1 = __builtin_shufflevector(b1l, b1h, 0,1,2,3,4,5,6,7,8,9,10,11,12,13,14,15);
    acc0 = __builtin_amdgcn_wmma_f32_16x16x32_bf16(false, a, false, b0,
                                                   (short)0, acc0, false, false);
    acc1 = __builtin_amdgcn_wmma_f32_16x16x32_bf16(false, a, false, b1,
                                                   (short)0, acc1, false, false);
  }
  // D layout: lane holds col N = lane&15; VGPR r holds row M = r + 8*(lane>>4)
  const int c  = lane & 15;
  const int hi = lane >> 4;
#pragma unroll
  for (int r = 0; r < 8; ++r) {
    const int M = r + hi * 8;
    xl[(size_t)(mb * 16 + M) * HC + nb0 * 16 + c] = acc0[r];
    xl[(size_t)(mb * 16 + M) * HC + nb1 * 16 + c] = acc1[r];
  }
}

// ---------------------------------------------------------------------------
// Kernel 2: per-(node,head) attention dots alpha_l / alpha_r
// ---------------------------------------------------------------------------
__global__ void k_node_alpha(const float* __restrict__ xl,
                             const float* __restrict__ att_l,
                             const float* __restrict__ att_r,
                             float* __restrict__ al, float* __restrict__ ar) {
  int idx = blockIdx.x * blockDim.x + threadIdx.x;
  if (idx >= N_NODES * NHEAD) return;
  const int n = idx >> 2, h = idx & 3;
  const float* xp  = xl + (size_t)n * HC + h * DOUT;
  const float* alp = att_l + h * DOUT;
  const float* arp = att_r + h * DOUT;
  float sl = 0.0f, sr = 0.0f;
#pragma unroll 8
  for (int c = 0; c < DOUT; ++c) { float v = xp[c]; sl += v * alp[c]; sr += v * arp[c]; }
  al[idx] = sl; ar[idx] = sr;
}

// ---------------------------------------------------------------------------
// Kernel 3: fused edge projection: alpha_e [E,4] and e_out [E,16]
// e never materialized (saves 205 MB). W_e (4 KB) in LDS.
// ---------------------------------------------------------------------------
__global__ __launch_bounds__(256)
void k_edge_proj(const float* __restrict__ eattr, const float* __restrict__ We,
                 const float* __restrict__ att_e, const float* __restrict__ ebias,
                 float* __restrict__ ae, float* __restrict__ eout) {
  __shared__ float sW[64 * 16];
  __shared__ float sA[64];
  __shared__ float sB[16];
  for (int i = threadIdx.x; i < 64 * 16; i += blockDim.x) sW[i] = We[i];
  for (int i = threadIdx.x; i < 64;      i += blockDim.x) sA[i] = att_e[i];
  for (int i = threadIdx.x; i < 16;      i += blockDim.x) sB[i] = ebias[i];
  __syncthreads();

  const int e = blockIdx.x * blockDim.x + threadIdx.x;
  if (e >= N_EDGES) return;

  float v[16];
  const float4* ep = (const float4*)(eattr + (size_t)e * EIN);
#pragma unroll
  for (int q = 0; q < 4; ++q) {
    float4 t = ep[q];
    v[q * 4 + 0] = t.x; v[q * 4 + 1] = t.y; v[q * 4 + 2] = t.z; v[q * 4 + 3] = t.w;
  }

  float aeh[NHEAD] = {0.f, 0.f, 0.f, 0.f};
  float* op = eout + (size_t)e * EOUT;
#pragma unroll
  for (int c = 0; c < EOUT; ++c) {
    float mx = -3.0e38f;
#pragma unroll
    for (int h = 0; h < NHEAD; ++h) {
      const int j = h * 16 + c;
      const float* wr = sW + j * 16;
      float acc = 0.0f;
#pragma unroll
      for (int k = 0; k < 16; ++k) acc = fmaf(v[k], wr[k], acc);
      aeh[h] += acc * sA[j];
      mx = fmaxf(mx, acc);
    }
    op[c] = fmaxf(mx + sB[c], 0.0f);
  }
#pragma unroll
  for (int h = 0; h < NHEAD; ++h) ae[(size_t)e * NHEAD + h] = aeh[h];
}

// ---------------------------------------------------------------------------
// Kernel 4: logits + leaky-relu + segment max (float atomic max)
// ---------------------------------------------------------------------------
__global__ void k_logits_max(const int* __restrict__ ei,
                             const float* __restrict__ al,
                             const float* __restrict__ ar,
                             float* __restrict__ lg, float* __restrict__ m) {
  const int e = blockIdx.x * blockDim.x + threadIdx.x;
  if (e >= N_EDGES) return;
  const int src = ei[e], dst = ei[N_EDGES + e];
#pragma unroll
  for (int h = 0; h < NHEAD; ++h) {
    float l = al[src * NHEAD + h] + ar[dst * NHEAD + h] + lg[(size_t)e * NHEAD + h];
    l = (l >= 0.0f) ? l : NEG_SLOPE * l;
    lg[(size_t)e * NHEAD + h] = l;
    atomicMaxF(&m[dst * NHEAD + h], l);
  }
}

// ---------------------------------------------------------------------------
// Kernel 5: exp(logit - m[dst]) + segment sum
// ---------------------------------------------------------------------------
__global__ void k_exp_sum(const int* __restrict__ ei, float* __restrict__ lg,
                          const float* __restrict__ m, float* __restrict__ s) {
  const int e = blockIdx.x * blockDim.x + threadIdx.x;
  if (e >= N_EDGES) return;
  const int dst = ei[N_EDGES + e];
#pragma unroll
  for (int h = 0; h < NHEAD; ++h) {
    float ex = __expf(lg[(size_t)e * NHEAD + h] - m[dst * NHEAD + h]);
    lg[(size_t)e * NHEAD + h] = ex;
    atomicAdd(&s[dst * NHEAD + h], ex);
  }
}

// ---------------------------------------------------------------------------
// Kernel 6: weighted scatter: accum[dst] += alpha * xl[src]
// 64 threads per edge, 4 contiguous channels each (xl + accum L2-resident)
// ---------------------------------------------------------------------------
__global__ __launch_bounds__(256)
void k_scatter(const int* __restrict__ ei, const float* __restrict__ lg,
               const float* __restrict__ s, const float* __restrict__ xl,
               float* __restrict__ accum) {
  const int gid = blockIdx.x * blockDim.x + threadIdx.x;
  const int e = gid >> 6;
  if (e >= N_EDGES) return;
  const int t = gid & 63;
  const int c = t * 4;
  const int h = c >> 6;
  const int src = ei[e], dst = ei[N_EDGES + e];
  const float alpha = lg[(size_t)e * NHEAD + h] / (s[dst * NHEAD + h] + 1e-16f);
  const float4 xv = *(const float4*)(xl + (size_t)src * HC + c);
  float* ap = accum + (size_t)dst * HC + c;
  atomicAdd(ap + 0, alpha * xv.x);
  atomicAdd(ap + 1, alpha * xv.y);
  atomicAdd(ap + 2, alpha * xv.z);
  atomicAdd(ap + 3, alpha * xv.w);
}

// ---------------------------------------------------------------------------
// Kernel 7: node epilogue: max over heads + bias + relu
// ---------------------------------------------------------------------------
__global__ void k_node_out(const float* __restrict__ accum,
                           const float* __restrict__ nbias,
                           float* __restrict__ out) {
  const int idx = blockIdx.x * blockDim.x + threadIdx.x;
  if (idx >= N_NODES * DOUT) return;
  const int n = idx >> 6, c = idx & 63;
  const float* ap = accum + (size_t)n * HC;
  float mx = ap[c];
#pragma unroll
  for (int h = 1; h < NHEAD; ++h) mx = fmaxf(mx, ap[h * DOUT + c]);
  out[idx] = fmaxf(mx + nbias[c], 0.0f);
}

// ---------------------------------------------------------------------------
extern "C" void kernel_launch(void* const* d_in, const int* in_sizes, int n_in,
                              void* d_out, int out_size, void* d_ws, size_t ws_size,
                              hipStream_t stream) {
  const float* x     = (const float*)d_in[0];
  const float* eattr = (const float*)d_in[1];
  const int*   ei    = (const int*)d_in[2];
  const float* Wl    = (const float*)d_in[3];
  const float* We    = (const float*)d_in[4];
  const float* att_l = (const float*)d_in[5];
  const float* att_r = (const float*)d_in[6];
  const float* att_e = (const float*)d_in[7];
  const float* nbias = (const float*)d_in[8];
  const float* ebias = (const float*)d_in[9];

  float* out_nodes = (float*)d_out;                          // [N,64]
  float* out_edges = out_nodes + (size_t)N_NODES * DOUT;     // [E,16]

  // workspace layout: fp32 region (~118.4 MB) then bf16 region (~12.9 MB)
  float* xl    = (float*)d_ws;                               // N*256
  float* al    = xl + (size_t)N_NODES * HC;                  // N*4
  float* ar    = al + (size_t)N_NODES * NHEAD;               // N*4
  float* ae    = ar + (size_t)N_NODES * NHEAD;               // E*4
  float* m     = ae + (size_t)N_EDGES * NHEAD;               // N*4
  float* s     = m  + (size_t)N_NODES * NHEAD;               // N*4
  float* accum = s  + (size_t)N_NODES * NHEAD;               // N*256
  __bf16* xb   = (__bf16*)(accum + (size_t)N_NODES * HC);    // N*128 bf16
  __bf16* wb   = xb + (size_t)XB_ELEMS;                      // 256*128 bf16

  const int B = 256;
  k_init<<<(N_NODES * HC + B - 1) / B, B, 0, stream>>>(m, s, accum);
  k_prep_bf16<<<(XB_ELEMS + WB_ELEMS + B - 1) / B, B, 0, stream>>>(x, Wl, xb, wb);
  k_node_proj_wmma<<<N_NODES / 16, B, 0, stream>>>(xb, wb, xl);
  k_edge_proj<<<(N_EDGES + B - 1) / B, B, 0, stream>>>(eattr, We, att_e, ebias, ae, out_edges);
  k_node_alpha<<<(N_NODES * NHEAD + B - 1) / B, B, 0, stream>>>(xl, att_l, att_r, al, ar);
  k_logits_max<<<(N_EDGES + B - 1) / B, B, 0, stream>>>(ei, al, ar, ae, m);
  k_exp_sum<<<(N_EDGES + B - 1) / B, B, 0, stream>>>(ei, ae, m, s);
  k_scatter<<<(N_EDGES * 64) / B, B, 0, stream>>>(ei, ae, s, xl, accum);
  k_node_out<<<(N_NODES * DOUT + B - 1) / B, B, 0, stream>>>(accum, nbias, out_nodes);
}
